// BConv2d_33423435498262
// MI455X (gfx1250) — 8-bit matrix operand order repaired and checked by reading, compiled, not run on hardware
//
#include <hip/hip_runtime.h>
#include <stdint.h>

typedef __attribute__((ext_vector_type(8))) int v8i;

#define NB    32
#define CIN   256
#define COUT  256
#define HH    56
#define WW    56
#define PIX   (HH*WW)
#define XB_H  58
#define XB_W  58
#define XB_NSTRIDE (XB_H*XB_W*CIN)
#define XB_BYTES ((size_t)NB*XB_NSTRIDE)
#define WB_BYTES (9*COUT*CIN)

#define LROW 144
#define LDS_BYTES (5 * 58 * LROW)

__global__ __launch_bounds__(256) void zero_ws_kernel(int4* p, int n16) {
  int i = blockIdx.x * 256 + threadIdx.x;
  if (i < n16) p[i] = make_int4(0, 0, 0, 0);
}

__global__ __launch_bounds__(256) void binarize_x_kernel(
    const float* __restrict__ x, int8_t* __restrict__ xb) {
  __shared__ int8_t lds[HH * 260];
  const int h = blockIdx.x;
  const int n = blockIdx.y;
  const int t = threadIdx.x;
  const int wq = t & 63;
  const int cq = t >> 6;

  for (int it = 0; it < 64; ++it) {
    const int ci = it * 4 + cq;
    if (wq < HH) {
      float v = x[(((size_t)n * CIN + ci) * HH + h) * WW + wq];
      lds[wq * 260 + ci] = (v >= 0.0f) ? (int8_t)1 : (int8_t)-1;
    }
  }
  __syncthreads();

  int8_t* outbase = xb + ((size_t)((n * XB_H + h + 1) * XB_W + 1)) * CIN;
  for (int i = t; i < HH * 64; i += 256) {
    const int w  = i >> 6;
    const int dw = i & 63;
    int val = *(const int*)&lds[w * 260 + dw * 4];
    *(int*)(outbase + (size_t)w * CIN + dw * 4) = val;
  }
}

__global__ __launch_bounds__(256) void binarize_w_kernel(
    const float* __restrict__ w, int8_t* __restrict__ wb) {
  int idx = blockIdx.x * 256 + threadIdx.x;
  if (idx >= WB_BYTES) return;
  const int ci  = idx & 255;
  const int co  = (idx >> 8) & 255;
  const int tap = idx >> 16;
  float v = w[((size_t)(co * CIN + ci)) * 9 + tap];
  wb[idx] = (v >= 0.0f) ? (int8_t)1 : (int8_t)-1;
}

__global__ __launch_bounds__(256) void bconv_wmma_kernel(
    const int8_t* __restrict__ xb, const int8_t* __restrict__ wb,
    const float* __restrict__ bias, float* __restrict__ out) {
  __shared__ int8_t slds[LDS_BYTES];

  const int tid  = threadIdx.x;
  const int lane = tid & 31;
  const int cg   = tid >> 5;
  const int bk   = blockIdx.x;
  const int pt4  = bk % 49;
  const int n    = bk / 49;
  const int hi   = lane >> 4;
  const int ln   = lane & 15;
  const int co0  = cg * 32;

  const int P0 = pt4 * 64;
  const int h0 = P0 / WW;
  const int h1 = (P0 + 63) / WW;
  const int nrows = h1 - h0 + 3;

  int pp[4], lbase[4];
#pragma unroll
  for (int nt = 0; nt < 4; ++nt) {
    int p  = P0 + nt * 16 + ln;
    pp[nt] = p;
    int hhp = p / WW;
    int wwp = p % WW;
    lbase[nt] = ((hhp - h0) * XB_W + wwp) * LROW + hi * 16;
  }
  const int8_t* xrow = xb + (size_t)n * XB_NSTRIDE + (size_t)h0 * XB_W * CIN;

  const v8i vzero = {0, 0, 0, 0, 0, 0, 0, 0};
  v8i acc[2][4];
#pragma unroll
  for (int mt = 0; mt < 2; ++mt)
#pragma unroll
    for (int nt = 0; nt < 4; ++nt) acc[mt][nt] = vzero;

  for (int khalf = 0; khalf < 2; ++khalf) {
    {
      const int total = nrows * XB_W * 8;
      const int8_t* src0 = xrow + khalf * 128;
      for (int i = tid; i < total; i += 256) {
        const int pos = i >> 3;
        const int q   = i & 7;
        *(int4*)&slds[pos * LROW + q * 16] =
            *(const int4*)(src0 + (size_t)pos * CIN + q * 16);
      }
    }
    __syncthreads();

    for (int kh = 0; kh < 3; ++kh) {
      for (int kw = 0; kw < 3; ++kw) {
        const int tap    = kh * 3 + kw;
        const int tapoff = (kh * XB_W + kw) * LROW;
        const int8_t* aBase0 =
            wb + ((size_t)(tap * COUT + co0 + ln) << 8) + khalf * 128;
        const int8_t* aBase1 = aBase0 + (16 << 8);

#pragma unroll
        for (int ks2 = 0; ks2 < 2; ++ks2) {
          const int k0 = ks2 * 64;
          v8i A[2];
          {
            const int8_t* ab = aBase0 + k0 + hi * 16;
            int4 q0 = *(const int4*)(ab);
            int4 q1 = *(const int4*)(ab + 32);
            v8i a = {q0.x, q0.y, q0.z, q0.w, q1.x, q1.y, q1.z, q1.w};
            A[0] = a;
          }
          {
            const int8_t* ab = aBase1 + k0 + hi * 16;
            int4 q0 = *(const int4*)(ab);
            int4 q1 = *(const int4*)(ab + 32);
            v8i a = {q0.x, q0.y, q0.z, q0.w, q1.x, q1.y, q1.z, q1.w};
            A[1] = a;
          }
#pragma unroll
          for (int nt = 0; nt < 4; ++nt) {
            const int off = lbase[nt] + tapoff + k0;
            int4 r0 = *(const int4*)&slds[off];
            int4 r1 = *(const int4*)&slds[off + 32];
            v8i B = {r0.x, r0.y, r0.z, r0.w, r1.x, r1.y, r1.z, r1.w};
#pragma unroll
            for (int mt = 0; mt < 2; ++mt) {
              acc[mt][nt] = __builtin_amdgcn_wmma_i32_16x16x64_iu8(
                  true, A[mt], true, B, acc[mt][nt], false, false);
            }
          }
        }
      }
    }
    __syncthreads();
  }

#pragma unroll
  for (int mt = 0; mt < 2; ++mt) {
#pragma unroll
    for (int r = 0; r < 8; ++r) {
      const int co = co0 + mt * 16 + hi * 8 + r;
      const float bv = bias[co];
      float* orow = out + ((size_t)(n * COUT + co)) * PIX;
#pragma unroll
      for (int nt = 0; nt < 4; ++nt) {
        orow[pp[nt]] = (float)acc[mt][nt][r] + bv;
      }
    }
  }
}

extern "C" void kernel_launch(void* const* d_in, const int* in_sizes, int n_in,
                              void* d_out, int out_size, void* d_ws, size_t ws_size,
                              hipStream_t stream) {
  (void)in_sizes; (void)n_in; (void)out_size; (void)ws_size;
  const float* x    = (const float*)d_in[0];
  const float* w    = (const float*)d_in[1];
  const float* bias = (const float*)d_in[2];
  float* out        = (float*)d_out;

  int8_t* xb   = (int8_t*)d_ws;
  int8_t* wbuf = xb + XB_BYTES;

  const int n16 = (int)(XB_BYTES / 16);
  zero_ws_kernel<<<(n16 + 255) / 256, 256, 0, stream>>>((int4*)xb, n16);

  dim3 gbx(HH, NB);
  binarize_x_kernel<<<gbx, 256, 0, stream>>>(x, xb);

  binarize_w_kernel<<<(WB_BYTES + 255) / 256, 256, 0, stream>>>(w, wbuf);

  bconv_wmma_kernel<<<NB * 49, 256, 0, stream>>>(xb, wbuf, bias, out);
}
